// PosActions_7430293422107
// MI455X (gfx1250) — compile-verified, matrix operand order checked by reading, not run on hardware
//
#include <hip/hip_runtime.h>
#include <hip/hip_bf16.h>
#include <math.h>

typedef __attribute__((ext_vector_type(16))) __bf16 v16bf;
typedef __attribute__((ext_vector_type(8)))  __bf16 v8bf;
typedef __attribute__((ext_vector_type(8)))  float  v8f;

#define DD   2048
#define BB   1024
#define PP   361
#define BM   64
#define BN   64
#define BK   32

// Fragment-ordered LDS tile: 4 subtiles (16 rows each) x 32 lanes x 32B.
// Lane l of a subtile holds row (l&15), K-halfbase kb=(l>>4)*8, with
// elements i: K = (i/8)*16 + kb + (i%8)  ->  two contiguous 8-elem K runs.
// Byte address of (subtile s, lane, group g) = s*1024 + lane*32 + g*16.

// ---------------------------------------------------------------------------
// Kernel 1: kp = pf @ W_kp + b_kp, stored transposed bf16: kpT[d][p]
// ---------------------------------------------------------------------------
__global__ __launch_bounds__(256) void kp_kernel(const float* __restrict__ pf,
                                                 const float* __restrict__ Wkp,
                                                 const float* __restrict__ bkp,
                                                 __bf16* __restrict__ kpT) {
    int id = blockIdx.x * 256 + threadIdx.x;
    if (id >= PP * DD) return;
    int p = id >> 11;
    int d = id & (DD - 1);
    float s = bkp[d];
    const float* pr = pf + p * 64;
#pragma unroll 8
    for (int j = 0; j < 64; ++j) s += pr[j] * Wkp[j * DD + d];
    kpT[d * PP + p] = (__bf16)s;
}

// ---------------------------------------------------------------------------
// Kernel 2: fused kx / q GEMM (A = x shared), fragment-ordered LDS staging.
// ---------------------------------------------------------------------------
__global__ __launch_bounds__(256) void kxq_gemm(const float* __restrict__ x,
                                                const float* __restrict__ Wkx,
                                                const float* __restrict__ bias_kx,
                                                const float* __restrict__ Wq,
                                                const float* __restrict__ bias_q,
                                                __bf16* __restrict__ qout,
                                                float* __restrict__ rowdot) {
    __shared__ alignas(32) __bf16 AsF[4 * 512];   // 4KB  (M subtiles)
    __shared__ alignas(32) __bf16 BkF[4 * 512];   // 4KB  (N subtiles)
    __shared__ alignas(32) __bf16 BqF[4 * 512];   // 4KB

    const int tid  = threadIdx.x;
    const int lane = tid & 31;
    const int wid  = tid >> 5;
    const int m0 = blockIdx.y * BM;
    const int n0 = blockIdx.x * BN;

    const int sm  = wid >> 1;             // M subtile (mbase = sm*16)
    const int sn0 = (wid & 1) * 2;        // first of two N subtiles
    const int nb0 = sn0 * 16, nb1 = nb0 + 16;

    // A staging: thread -> (row, 8-long K run); 4 threads cover one 128B row chunk
    const int a_m    = tid >> 2;          // 0..63
    const int a_kc   = (tid & 3) * 8;     // 0,8,16,24
    const int a_s    = a_m >> 4;
    const int a_lane = (a_m & 15) + ((a_kc & 8) ? 16 : 0);
    const int a_g    = a_kc >> 4;
    __bf16* a_dst = AsF + a_s * 512 + a_lane * 16 + a_g * 8;
    const float* a_src = x + (size_t)(m0 + a_m) * DD + a_kc;

    // B staging: thread -> (one n, 8-long K run); lanes have consecutive n
    const int b_n    = tid & 63;
    const int b_kc   = (tid >> 6) * 8;    // 0,8,16,24
    const int b_s    = b_n >> 4;
    const int b_lane = (b_n & 15) + ((b_kc & 8) ? 16 : 0);
    const int b_g    = b_kc >> 4;
    const int b_off  = b_s * 512 + b_lane * 16 + b_g * 8;
    const float* wk_src = Wkx + (size_t)b_kc * DD + n0 + b_n;
    const float* wq_src = Wq  + (size_t)b_kc * DD + n0 + b_n;

    v8f ck0 = {}, ck1 = {}, cq0 = {}, cq1 = {};

    for (int k0 = 0; k0 < DD; k0 += BK) {
        // ---- stage A (vector global loads, one 16B LDS store)
        const float4 x0 = *(const float4*)(a_src + k0);
        const float4 x1 = *(const float4*)(a_src + k0 + 4);
        v8bf av;
        av[0] = (__bf16)x0.x; av[1] = (__bf16)x0.y; av[2] = (__bf16)x0.z; av[3] = (__bf16)x0.w;
        av[4] = (__bf16)x1.x; av[5] = (__bf16)x1.y; av[6] = (__bf16)x1.z; av[7] = (__bf16)x1.w;
        *(v8bf*)a_dst = av;

        // ---- stage both B tiles (k-strided, lane-coalesced loads; 16B stores)
        const float* wkp = wk_src + (size_t)k0 * DD;
        const float* wqp = wq_src + (size_t)k0 * DD;
        v8bf bk, bq;
#pragma unroll
        for (int j = 0; j < 8; ++j) {
            bk[j] = (__bf16)wkp[(size_t)j * DD];
            bq[j] = (__bf16)wqp[(size_t)j * DD];
        }
        *(v8bf*)(BkF + b_off) = bk;
        *(v8bf*)(BqF + b_off) = bq;
        __syncthreads();

        // ---- fragments: one aligned 32B load each
        const v16bf fa  = *(const v16bf*)(AsF + sm * 512 + lane * 16);
        const v16bf fk0 = *(const v16bf*)(BkF + sn0 * 512 + lane * 16);
        const v16bf fk1 = *(const v16bf*)(BkF + (sn0 + 1) * 512 + lane * 16);
        const v16bf fq0 = *(const v16bf*)(BqF + sn0 * 512 + lane * 16);
        const v16bf fq1 = *(const v16bf*)(BqF + (sn0 + 1) * 512 + lane * 16);

        ck0 = __builtin_amdgcn_wmma_f32_16x16x32_bf16(false, fa, false, fk0, (short)0, ck0, false, false);
        ck1 = __builtin_amdgcn_wmma_f32_16x16x32_bf16(false, fa, false, fk1, (short)0, ck1, false, false);
        cq0 = __builtin_amdgcn_wmma_f32_16x16x32_bf16(false, fa, false, fq0, (short)0, cq0, false, false);
        cq1 = __builtin_amdgcn_wmma_f32_16x16x32_bf16(false, fa, false, fq1, (short)0, cq1, false, false);
        __syncthreads();
    }

    // ---- epilogue: +bias, store q bf16, rowdot[m] += sum_n kx*q
    const int half = lane >> 4;
    const int nl   = lane & 15;
    const int gn0 = n0 + nb0 + nl;
    const int gn1 = n0 + nb1 + nl;
    const float bk0 = bias_kx[gn0], bk1 = bias_kx[gn1];
    const float bq0 = bias_q [gn0], bq1 = bias_q [gn1];
#pragma unroll
    for (int r = 0; r < 8; ++r) {
        const int gm = m0 + sm * 16 + r + half * 8;
        const float kx0 = ck0[r] + bk0;
        const float kx1 = ck1[r] + bk1;
        const float qv0 = cq0[r] + bq0;
        const float qv1 = cq1[r] + bq1;
        qout[(size_t)gm * DD + gn0] = (__bf16)qv0;
        qout[(size_t)gm * DD + gn1] = (__bf16)qv1;
        float prod = kx0 * qv0 + kx1 * qv1;
        prod += __shfl_xor(prod, 1, 32);    // xor 1,2,4,8: stays within 16-lane half
        prod += __shfl_xor(prod, 2, 32);
        prod += __shfl_xor(prod, 4, 32);
        prod += __shfl_xor(prod, 8, 32);
        if (nl == 0) atomicAdd(&rowdot[gm], prod);
    }
}

// ---------------------------------------------------------------------------
// Kernel 3: dots = (q @ kpT + rowdot) / sqrt(D).  A staged via async-to-LDS.
// ---------------------------------------------------------------------------
__global__ __launch_bounds__(256) void dots_gemm(const __bf16* __restrict__ qbf,
                                                 const __bf16* __restrict__ kpT,
                                                 const float* __restrict__ rowdot,
                                                 float* __restrict__ dots) {
    __shared__ alignas(32) __bf16 AsF[4 * 512];
    __shared__ alignas(32) __bf16 BsF[4 * 512];

    const int tid  = threadIdx.x;
    const int lane = tid & 31;
    const int wid  = tid >> 5;
    const int m0 = blockIdx.y * BM;
    const int n0 = blockIdx.x * BN;

    const int sm  = wid >> 1;
    const int sn0 = (wid & 1) * 2;
    const int nb0 = sn0 * 16, nb1 = nb0 + 16;

    // A: bf16 source -> direct async global->LDS 16B copy into fragment slot
    const int a_m    = tid >> 2;
    const int a_kc   = (tid & 3) * 8;
    const int a_s    = a_m >> 4;
    const int a_lane = (a_m & 15) + ((a_kc & 8) ? 16 : 0);
    const int a_g    = a_kc >> 4;
    const unsigned a_lds = (unsigned)(uintptr_t)(AsF + a_s * 512 + a_lane * 16 + a_g * 8);
    const __bf16* a_src = qbf + (size_t)(m0 + a_m) * DD + a_kc;

    // B: from kpT [K][P] bf16, zero-pad n >= 361
    const int b_n    = tid & 63;
    const int b_kc   = (tid >> 6) * 8;
    const int b_s    = b_n >> 4;
    const int b_lane = (b_n & 15) + ((b_kc & 8) ? 16 : 0);
    const int b_g    = b_kc >> 4;
    const int b_off  = b_s * 512 + b_lane * 16 + b_g * 8;
    const int gn     = n0 + b_n;
    const bool nvalid = (gn < PP);
    const __bf16* b_src = kpT + (size_t)b_kc * PP + (nvalid ? gn : 0);

    v8f c0 = {}, c1 = {};

    for (int k0 = 0; k0 < DD; k0 += BK) {
        // ---- async copy A fragment run: global(16B) -> LDS, ASYNCcnt-tracked
        const unsigned long long ga = (unsigned long long)(uintptr_t)(a_src + k0);
        asm volatile("global_load_async_to_lds_b128 %0, %1, off"
                     :: "v"(a_lds), "v"(ga) : "memory");

        // ---- stage B (lane-coalesced bf16 loads)
        const __bf16* bp = b_src + (size_t)k0 * PP;
        v8bf bv;
#pragma unroll
        for (int j = 0; j < 8; ++j)
            bv[j] = nvalid ? bp[(size_t)j * PP] : (__bf16)0.0f;
        *(v8bf*)(BsF + b_off) = bv;

        asm volatile("s_wait_asynccnt 0" ::: "memory");
        __syncthreads();

        const v16bf fa  = *(const v16bf*)(AsF + sm * 512 + lane * 16);
        const v16bf fb0 = *(const v16bf*)(BsF + sn0 * 512 + lane * 16);
        const v16bf fb1 = *(const v16bf*)(BsF + (sn0 + 1) * 512 + lane * 16);

        c0 = __builtin_amdgcn_wmma_f32_16x16x32_bf16(false, fa, false, fb0, (short)0, c0, false, false);
        c1 = __builtin_amdgcn_wmma_f32_16x16x32_bf16(false, fa, false, fb1, (short)0, c1, false, false);
        __syncthreads();
    }

    const float scale = 0.02209708691207961f;  // 1/sqrt(2048)
    const int half = lane >> 4;
    const int nl   = lane & 15;
    const int gn0 = n0 + nb0 + nl;
    const int gn1 = n0 + nb1 + nl;
#pragma unroll
    for (int r = 0; r < 8; ++r) {
        const int gm = m0 + sm * 16 + r + half * 8;
        const float rd = rowdot[gm];
        if (gn0 < PP) dots[(size_t)gm * PP + gn0] = (c0[r] + rd) * scale;
        if (gn1 < PP) dots[(size_t)gm * PP + gn1] = (c1[r] + rd) * scale;
    }
}

// ---------------------------------------------------------------------------
// Kernel 4: row-wise log_softmax over 361 entries.
// ---------------------------------------------------------------------------
__global__ __launch_bounds__(512) void logsoftmax_kernel(const float* __restrict__ dots,
                                                         float* __restrict__ out) {
    __shared__ float red_max[16];
    __shared__ float red_sum[16];
    const int row = blockIdx.x;
    const int tid = threadIdx.x;
    const float v = (tid < PP) ? dots[(size_t)row * PP + tid] : -INFINITY;

    float m = v;
#pragma unroll
    for (int s = 16; s > 0; s >>= 1) m = fmaxf(m, __shfl_xor(m, s, 32));
    if ((tid & 31) == 0) red_max[tid >> 5] = m;
    __syncthreads();
    if (tid < 32) {
        float t = (tid < 16) ? red_max[tid] : -INFINITY;
#pragma unroll
        for (int s = 8; s > 0; s >>= 1) t = fmaxf(t, __shfl_xor(t, s, 32));
        if (tid == 0) red_max[0] = t;
    }
    __syncthreads();
    m = red_max[0];

    float e = (tid < PP) ? __expf(v - m) : 0.0f;
#pragma unroll
    for (int s = 16; s > 0; s >>= 1) e += __shfl_xor(e, s, 32);
    if ((tid & 31) == 0) red_sum[tid >> 5] = e;
    __syncthreads();
    if (tid < 32) {
        float t = (tid < 16) ? red_sum[tid] : 0.0f;
#pragma unroll
        for (int s = 8; s > 0; s >>= 1) t += __shfl_xor(t, s, 32);
        if (tid == 0) red_sum[0] = t;
    }
    __syncthreads();
    const float lse = m + logf(red_sum[0]);
    if (tid < PP) out[(size_t)row * PP + tid] = v - lse;
}

// ---------------------------------------------------------------------------
extern "C" void kernel_launch(void* const* d_in, const int* in_sizes, int n_in,
                              void* d_out, int out_size, void* d_ws, size_t ws_size,
                              hipStream_t stream) {
    const float* x    = (const float*)d_in[0];
    const float* p    = (const float*)d_in[1];   // [19,19,64] contiguous == pf [361,64]
    const float* W_kp = (const float*)d_in[2];
    const float* b_kp = (const float*)d_in[3];
    const float* W_kx = (const float*)d_in[4];
    const float* b_kx = (const float*)d_in[5];
    const float* W_q  = (const float*)d_in[6];
    const float* b_q  = (const float*)d_in[7];
    float* out = (float*)d_out;

    char* ws = (char*)d_ws;
    __bf16* qbf = (__bf16*)ws;                                   // 1024*2048 bf16
    size_t off = (size_t)BB * DD * sizeof(__bf16);
    __bf16* kpT = (__bf16*)(ws + off);                           // 2048*361 bf16
    off += (((size_t)DD * PP * sizeof(__bf16)) + 255) & ~(size_t)255;
    float* rowdot = (float*)(ws + off);                          // 1024 f32
    off += 4096;
    float* dots = (float*)(ws + off);                            // 1024*361 f32

    hipMemsetAsync(rowdot, 0, BB * sizeof(float), stream);

    kp_kernel<<<(PP * DD + 255) / 256, 256, 0, stream>>>(p, W_kp, b_kp, kpT);
    kxq_gemm<<<dim3(DD / BN, BB / BM), 256, 0, stream>>>(x, W_kx, b_kx, W_q, b_q, qbf, rowdot);
    dots_gemm<<<dim3((PP + BN - 1) / BN, BB / BM), 256, 0, stream>>>(qbf, kpT, rowdot, dots);
    logsoftmax_kernel<<<BB, 512, 0, stream>>>(dots, out);
}
